// GraphMultiHeadAttentionLayer_11501922419375
// MI455X (gfx1250) — matrix operand order repaired and checked by reading, not compiled, not run on hardware
//
#include <hip/hip_runtime.h>
#include <hip/hip_bf16.h>
#include <math.h>

// ---------------------------------------------------------------------------
// GraphMultiHeadAttention for MI455X (gfx1250, wave32, WMMA 16x16x32 f16).
//
//   t1 = relu(nodes@nW1+b)        [N,64]   (WMMA, KD=128)
//   h  = relu(t1@nW2+b)           [N,64]   (WMMA, KD=64)
//   q  = h@Wq[hd]+bq  (all heads) [N,128]  (WMMA, headed weights)
//   k  = h@Wk[hd]+bk  (all heads) [N,128]  (WMMA, headed weights)
//   e1 = relu(edges@eW1+b)        [E,64]   (WMMA, KD=16 zero-padded)
//   ef = relu(e1@eW2+b)           [E,64]   (WMMA, KD=64)
//   ekey = ef@We[hd]+be           [E,128]  (WMMA, headed weights)
//   logits[e,hd] = scale*dot32(q[s], k[r]+ekey[e]); segment softmax via
//   atomic fmax (int trick) + atomicAdd; out[r,hd*64:] += wt*h[s].
//
// Weights are staged to LDS once per block, PRE-SWIZZLED into the CDNA5
// B-fragment layout, so each WMMA's B operand is two ds_load_b128.
// All M values are multiples of 16 -> no row guards, EXEC full at WMMA.
// ---------------------------------------------------------------------------

typedef __attribute__((ext_vector_type(16))) _Float16 v16h;
typedef __attribute__((ext_vector_type(8)))  float    v8f;

#define HDIM   64
#define KDIM   32
#define NHEADS 4

// ---------------- init: zero output, init segment max / denom --------------
__global__ void init_kernel(float* __restrict__ out, long nout,
                            float* __restrict__ mbuf, float* __restrict__ denom,
                            int nseg) {
  long t = (long)blockIdx.x * blockDim.x + threadIdx.x;
  if (t < nout) out[t] = 0.0f;
  if (t < nseg) { mbuf[t] = -3.402823466e+38f; denom[t] = 0.0f; }
}

// ---------------- WMMA GEMM: out[m, n] = act(X@W + bias) -------------------
// One wave = one 16-row M-tile, looping over 16-col N-tiles, K chunks of 32.
// Fragment layouts per CDNA5 ISA (16-bit A 16x32 / B 32x16 / f32 C 16x16):
//   A: lane(m=lane&15,kg=lane>>4), VGPR r<4: K=2r+8kg ; r>=4: K=16+2(r-4)+8kg
//   B: half i: K = (i&7)+8kg+16(i>>3), N = lane&15
//   C: vgpr v: M = v+8kg,  N = lane&15
// sW holds W pre-swizzled: sW[((nt*NK+kc)*32+lane)*16 + i] = B element.
// HEADED: W is [4][KD][32] per-head blocks mapped to output col blocks of 32.
template<int KD, int RELU, bool HEADED>
__global__ __launch_bounds__(256)
void gemm_wmma_kernel(const float* __restrict__ X, int ldx,
                      const float* __restrict__ W,
                      const float* __restrict__ bias,
                      float* __restrict__ out, int ldo,
                      int ntiles, int Mtiles) {
  constexpr int NK = (KD + 31) / 32;                // K chunks (1,2,4)
  __shared__ __align__(32) _Float16 sW[8192];       // 16KB, covers all configs

  const int tid  = threadIdx.x;
  const int Nout = ntiles * 16;
  const int nswz = ntiles * NK * 512;               // swizzled halves

  // ---- stage weights, pre-swizzled into B-fragment order (branch-free) ----
  for (int j = tid; j < nswz; j += 256) {
    int i  = j & 15;                                // half within lane chunk
    int l  = (j >> 4) & 31;                         // lane
    int fr = j >> 9;                                // fragment = nt*NK + kc
    int nt = fr / NK;                               // NK is power of 2
    int kc = fr & (NK - 1);
    int kg = l >> 4, m = l & 15;
    int k  = kc * 32 + (i & 7) + 8 * kg + ((i >> 3) << 4);
    int c  = nt * 16 + m;
    int kk = (KD % 32 != 0) ? ((k < KD) ? k : 0) : k;   // clamp (KD=16 only)
    int src = HEADED ? ((c >> 5) * (KD * 32) + kk * 32 + (c & 31))
                     : (kk * Nout + c);
    float wv = W[src];
    if (KD % 32 != 0) wv = (k < KD) ? wv : 0.0f;        // zero-pad K in [16,32)
    sW[j] = (_Float16)wv;
  }
  __syncthreads();

  const int lane  = tid & 31;
  const int wave  = tid >> 5;
  const int mtile = blockIdx.x * 8 + wave;          // wave-uniform
  if (mtile >= Mtiles) return;                      // uniform exit, EXEC full
  const int m  = lane & 15;
  const int kg = lane >> 4;

  // ---- A fragments: unconditional float2 (b64) loads, f32 -> f16 ----
  const float2* __restrict__ xrow =
      (const float2*)(X + (size_t)(mtile * 16 + m) * ldx);
  v16h afrag[NK];
  #pragma unroll
  for (int kc = 0; kc < NK; ++kc) {
    v16h a;
    #pragma unroll
    for (int r = 0; r < 8; ++r) {
      if ((KD % 32 != 0) && r >= 4) {               // KD=16: K in [16,32) pad
        a[2 * r]     = (_Float16)0.0f;
        a[2 * r + 1] = (_Float16)0.0f;
      } else {
        int pi = kc * 16 + ((r < 4) ? 0 : 8) + (r & 3) + 4 * kg;
        float2 xv = xrow[pi];
        a[2 * r]     = (_Float16)xv.x;
        a[2 * r + 1] = (_Float16)xv.y;
      }
    }
    afrag[kc] = a;
  }

  // ---- N-tiles: B fragment = 2x ds_load_b128, accumulate with WMMA ----
  for (int nt = 0; nt < ntiles; ++nt) {
    v8f cacc = {};
    #pragma unroll
    for (int kc = 0; kc < NK; ++kc) {
      const v16h b = *reinterpret_cast<const v16h*>(
          sW + (((nt * NK + kc) * 32 + lane) << 4));
      cacc = __builtin_amdgcn_wmma_f32_16x16x32_f16(false, afrag[kc], false, b,
                                                    (short)0, cacc, false, false);
    }
    const int c = nt * 16 + m;
    const float bv = bias[c];
    #pragma unroll
    for (int v = 0; v < 8; ++v) {
      float o = cacc[v] + bv;
      if (RELU) o = fmaxf(o, 0.0f);
      out[(size_t)(mtile * 16 + v + 8 * kg) * ldo + c] = o;
    }
  }
}

// ---------------- float atomic max via int/uint trick ----------------------
__device__ __forceinline__ void atomicMaxF(float* addr, float val) {
  if (val >= 0.0f) atomicMax((int*)addr, __float_as_int(val));
  else             atomicMin((unsigned int*)addr, __float_as_uint(val));
}

// ---------------- logits: 1 wave per edge, lane = K index ------------------
__global__ __launch_bounds__(256)
void logits_kernel(const int* __restrict__ senders, const int* __restrict__ receivers,
                   const float* __restrict__ q, const float* __restrict__ k,
                   const float* __restrict__ ekey,
                   float* __restrict__ logits, float* __restrict__ mbuf, int E) {
  const int lane = threadIdx.x & 31;
  const int e = (blockIdx.x * blockDim.x + threadIdx.x) >> 5;
  if (e >= E) return;
  const int s = senders[e], r = receivers[e];
  const float scale = 0.17677669529663687f;   // 1/sqrt(32)
  #pragma unroll
  for (int hd = 0; hd < NHEADS; ++hd) {
    const int col = hd * KDIM + lane;
    float qv = q[(size_t)s * 128 + col];
    float kv = k[(size_t)r * 128 + col] + ekey[(size_t)e * 128 + col];
    float p = qv * kv;
    #pragma unroll
    for (int off = 16; off > 0; off >>= 1) p += __shfl_xor(p, off, 32);
    if (lane == 0) {
      float lg = p * scale;
      logits[(size_t)e * NHEADS + hd] = lg;
      atomicMaxF(&mbuf[(size_t)r * NHEADS + hd], lg);
    }
  }
}

// ---------------- exp(logit - segmax) + segment denom ----------------------
__global__ void expsum_kernel(const int* __restrict__ receivers,
                              const float* __restrict__ logits,
                              const float* __restrict__ mbuf,
                              float* __restrict__ wbuf, float* __restrict__ denom,
                              int E) {
  int t = blockIdx.x * blockDim.x + threadIdx.x;
  if (t >= E * NHEADS) return;
  int e = t >> 2, hd = t & 3;
  int r = receivers[e];
  float w = __expf(logits[t] - mbuf[(size_t)r * NHEADS + hd]);
  wbuf[t] = w;
  atomicAdd(&denom[(size_t)r * NHEADS + hd], w);
}

// ---------------- weighted scatter-aggregate: 1 wave per edge --------------
__global__ __launch_bounds__(256)
void aggregate_kernel(const int* __restrict__ senders, const int* __restrict__ receivers,
                      const float* __restrict__ h, const float* __restrict__ wbuf,
                      const float* __restrict__ denom, float* __restrict__ out, int E) {
  const int lane = threadIdx.x & 31;
  const int e = (blockIdx.x * blockDim.x + threadIdx.x) >> 5;
  if (e >= E) return;
  const int s = senders[e], r = receivers[e];
  const float sv0 = h[(size_t)s * HDIM + lane * 2];
  const float sv1 = h[(size_t)s * HDIM + lane * 2 + 1];
  #pragma unroll
  for (int hd = 0; hd < NHEADS; ++hd) {
    float dnv = denom[(size_t)r * NHEADS + hd];
    float wt = wbuf[(size_t)e * NHEADS + hd] / fmaxf(dnv, 1e-9f);
    atomicAdd(&out[(size_t)r * (NHEADS * HDIM) + hd * HDIM + lane * 2],     wt * sv0);
    atomicAdd(&out[(size_t)r * (NHEADS * HDIM) + hd * HDIM + lane * 2 + 1], wt * sv1);
  }
}

// ---------------------------------------------------------------------------
template<int KD, int RELU, bool HEADED>
static void run_gemm(const float* X, int ldx, const float* W,
                     const float* bias, float* out, int ldo, int Nout,
                     int M, hipStream_t s) {
  int Mtiles = M / 16;                    // all M here are multiples of 16
  int blocks = (Mtiles + 7) / 8;
  gemm_wmma_kernel<KD, RELU, HEADED><<<blocks, 256, 0, s>>>(
      X, ldx, W, bias, out, ldo, Nout / 16, Mtiles);
}

extern "C" void kernel_launch(void* const* d_in, const int* in_sizes, int n_in,
                              void* d_out, int out_size, void* d_ws, size_t ws_size,
                              hipStream_t stream) {
  const float* nodes     = (const float*)d_in[0];
  const float* edges     = (const float*)d_in[1];
  const int*   senders   = (const int*)  d_in[2];
  const int*   receivers = (const int*)  d_in[3];
  const float* nW1 = (const float*)d_in[4];  const float* nb1 = (const float*)d_in[5];
  const float* nW2 = (const float*)d_in[6];  const float* nb2 = (const float*)d_in[7];
  const float* eW1 = (const float*)d_in[8];  const float* eb1 = (const float*)d_in[9];
  const float* eW2 = (const float*)d_in[10]; const float* eb2 = (const float*)d_in[11];
  const float* Wq  = (const float*)d_in[12]; const float* bq  = (const float*)d_in[13];
  const float* Wk  = (const float*)d_in[14]; const float* bk  = (const float*)d_in[15];
  const float* We  = (const float*)d_in[16]; const float* be  = (const float*)d_in[17];

  const int N = in_sizes[0] / 128;    // 50000
  const int E = in_sizes[1] / 16;     // 800000
  float* out = (float*)d_out;
  float* ws  = (float*)d_ws;

  // workspace layout (floats); region A reused stream-ordered: t1 -> e1 -> ekey
  size_t offA  = 0;                                  // E*128 region
  size_t offH  = offA  + (size_t)E * 128;            // h  [N,64]
  size_t offQ  = offH  + (size_t)N * HDIM;           // q  [N,128]
  size_t offK  = offQ  + (size_t)N * 128;            // k  [N,128]
  size_t offEF = offK  + (size_t)N * 128;            // ef [E,64]
  size_t offLG = offEF + (size_t)E * HDIM;           // logits [E,4]
  size_t offW  = offLG + (size_t)E * NHEADS;         // w      [E,4]
  size_t offM  = offW  + (size_t)E * NHEADS;         // segmax [N,4]
  size_t offD  = offM  + (size_t)N * NHEADS;         // denom  [N,4]

  float* t1    = ws + offA;   // [N,64]  (region A)
  float* e1    = ws + offA;   // [E,64]  (region A, after t1 dead)
  float* ekey  = ws + offA;   // [E,128] (region A, after e1 dead)
  float* h     = ws + offH;
  float* q     = ws + offQ;
  float* kbufv = ws + offK;
  float* ef    = ws + offEF;
  float* lg    = ws + offLG;
  float* wb    = ws + offW;
  float* mb    = ws + offM;
  float* dn    = ws + offD;

  // 1) init output + segment buffers (every call: deterministic)
  {
    long nout = (long)N * NHEADS * HDIM;
    long grid = (nout + 255) / 256;
    init_kernel<<<(int)grid, 256, 0, stream>>>(out, nout, mb, dn, N * NHEADS);
  }

  // 2) node MLP
  run_gemm<128, 1, false>(nodes, 128, nW1, nb1, t1, HDIM, HDIM, N, stream);
  run_gemm<64,  1, false>(t1, HDIM,  nW2, nb2, h,  HDIM, HDIM, N, stream);

  // 3) per-node q / k projections, all 4 heads in one launch (headed weights)
  run_gemm<64, 0, true>(h, HDIM, Wq, bq, q,     128, 128, N, stream);
  run_gemm<64, 0, true>(h, HDIM, Wk, bk, kbufv, 128, 128, N, stream);

  // 4) edge MLP + edge keys (all heads)
  run_gemm<16, 1, false>(edges, 16, eW1, eb1, e1, HDIM, HDIM, E, stream);
  run_gemm<64, 1, false>(e1, HDIM,  eW2, eb2, ef, HDIM, HDIM, E, stream);
  run_gemm<64, 0, true>(ef, HDIM, We, be, ekey, 128, 128, E, stream);

  // 5) attention logits + segment max (1 wave / edge)
  {
    int blocks = (E + 7) / 8;
    logits_kernel<<<blocks, 256, 0, stream>>>(senders, receivers, q, kbufv, ekey,
                                              lg, mb, E);
  }
  // 6) exp + denom
  {
    int blocks = (E * NHEADS + 255) / 256;
    expsum_kernel<<<blocks, 256, 0, stream>>>(receivers, lg, mb, wb, dn, E);
  }
  // 7) weighted scatter aggregation
  {
    int blocks = (E + 7) / 8;
    aggregate_kernel<<<blocks, 256, 0, stream>>>(senders, receivers, h, wb, dn, out, E);
  }
}
